// GatingCrossAttentionFusionResidual_38800734552606
// MI455X (gfx1250) — matrix operand order repaired and checked by reading, not compiled, not run on hardware
//
#include <hip/hip_runtime.h>
#include <hip/hip_bf16.h>

// ---------------------------------------------------------------------------
// GatingCrossAttentionFusionResidual for MI455X (gfx1250, wave32, WMMA).
// seq_len==1 MHA collapses to out_proj(v); Wq/Wk do not affect the output.
// Heavy math runs through v_wmma_f32_16x16x32_bf16 (f32 -> bf16 on staging,
// f32 accumulation). K-concat GEMMs (cross, gate) are fused via a 2-source
// register-accumulating loop; LDS is double-buffered (ping-pong) so global
// loads for tile k+1 overlap the WMMAs of tile k, one barrier per K-step,
// and the last iteration is peeled so the steady-state loop is branchless.
// ---------------------------------------------------------------------------

typedef __bf16 bf16_t;
typedef __attribute__((ext_vector_type(16))) __bf16 v16bf;
typedef __attribute__((ext_vector_type(8)))  __bf16 v8bf;
typedef __attribute__((ext_vector_type(8)))  float  v8f;

#define TILE_M 128
#define TILE_N 128
#define TILE_K 32
#define LDS_STRIDE 40  // 32 bf16 + 8 pad -> 80B row pitch, kills bank conflicts

union Frag16 { v16bf v; v8bf h[2]; };

__device__ __forceinline__ void load16(float f[16], const float* __restrict__ p) {
    const float4* s = (const float4*)p;
    *(float4*)&f[0]  = s[0];
    *(float4*)&f[4]  = s[1];
    *(float4*)&f[8]  = s[2];
    *(float4*)&f[12] = s[3];
}

__device__ __forceinline__ void cvt_store(bf16_t* __restrict__ dst, const float f[16]) {
    Frag16 p;
#pragma unroll
    for (int i = 0; i < 16; ++i) p.h[i >> 3][i & 7] = (bf16_t)f[i];
    ((v8bf*)dst)[0] = p.h[0];
    ((v8bf*)dst)[1] = p.h[1];
}

// C[M,2048] = sum_s A_s[M,K] @ W_s[:,K]^T + bias0 (+ bias1)
// A row-major (lda == K), W row-major with row pitch ldw (supports the
// gate_w[:, :F] / gate_w[:, F:] column-slab views via pointer offset + ldw).
template <int NSRC, bool HAS_B1>
__global__ __launch_bounds__(256)
void wmma_gemm(const float* __restrict__ A0, const float* __restrict__ W0,
               const float* __restrict__ A1, const float* __restrict__ W1,
               const float* __restrict__ bias0, const float* __restrict__ bias1,
               float* __restrict__ C, int K, int ldw)
{
    __shared__ bf16_t As[2][TILE_M * LDS_STRIDE];
    __shared__ bf16_t Ws[2][TILE_N * LDS_STRIDE];

    const int tid  = threadIdx.x;
    const int lane = tid & 31;
    const int wave = tid >> 5;
    const int l16  = lane & 15;
    const int hi   = lane >> 4;        // lane half (wave32)

    const int wm = wave & 3;           // 4 waves along M -> 32-row slab
    const int wn = wave >> 2;          // 2 waves along N -> 64-col slab

    const size_t blockM = (size_t)blockIdx.y * TILE_M;
    const int    blockN = blockIdx.x * TILE_N;
    const int    N      = 2048;

    // staging map: thread -> (row, 16-wide K chunk); 256 thr cover 128x32 tile
    const int srow = tid >> 1;
    const int skc  = (tid & 1) * 16;
    const int sdst = srow * LDS_STRIDE + skc;   // LDS offset for this thread

    v8f acc[2][4];
#pragma unroll
    for (int mf = 0; mf < 2; ++mf)
#pragma unroll
        for (int nf = 0; nf < 4; ++nf)
            acc[mf][nf] = (v8f){0.f, 0.f, 0.f, 0.f, 0.f, 0.f, 0.f, 0.f};

    // fragment-load + 8x WMMA for one staged K-tile in LDS buffer b
    auto compute_tile = [&](int b) {
        // A 16x32: lanes 0-15 hold K{0..7,16..23}, lanes 16-31 K{8..15,24..31}
        Frag16 afrag[2];
#pragma unroll
        for (int mf = 0; mf < 2; ++mf) {
            const bf16_t* p = &As[b][(wm * 32 + mf * 16 + l16) * LDS_STRIDE + hi * 8];
            afrag[mf].h[0] = *(const v8bf*)p;
            afrag[mf].h[1] = *(const v8bf*)(p + 16);
        }
        // B 32x16: col = lane%16; lanes 0-15 K{0..7,16..23}, lanes 16-31 K{8..15,24..31}
        Frag16 bfrag[4];
#pragma unroll
        for (int nf = 0; nf < 4; ++nf) {
            const bf16_t* p = &Ws[b][(wn * 64 + nf * 16 + l16) * LDS_STRIDE + hi * 8];
            bfrag[nf].h[0] = *(const v8bf*)p;
            bfrag[nf].h[1] = *(const v8bf*)(p + 16);
        }
#pragma unroll
        for (int mf = 0; mf < 2; ++mf)
#pragma unroll
            for (int nf = 0; nf < 4; ++nf)
                acc[mf][nf] = __builtin_amdgcn_wmma_f32_16x16x32_bf16(
                    false, afrag[mf].v, false, bfrag[nf].v,
                    (short)0, acc[mf][nf], false, false);
    };

#pragma unroll
    for (int s = 0; s < NSRC; ++s) {
        const float* Ab = (NSRC == 2 && s == 1) ? A1 : A0;
        const float* Wb = (NSRC == 2 && s == 1) ? W1 : W0;
        const float* aptr = Ab + (blockM + (size_t)srow) * (size_t)K + skc;
        const float* wptr = Wb + (size_t)(blockN + srow) * (size_t)ldw + skc;

        // prologue: stage tile 0 into LDS buffer 0
        float fa[16], fw[16];
        load16(fa, aptr);
        load16(fw, wptr);
        __syncthreads();                 // previous reads of As/Ws done
        cvt_store(&As[0][sdst], fa);
        cvt_store(&Ws[0][sdst], fw);

        const int nIter = K / TILE_K;
        int buf = 0;
        // steady state: branchless (last iteration peeled below)
        for (int it = 0; it < nIter - 1; ++it) {
            const int knext = (it + 1) * TILE_K;
            load16(fa, aptr + knext);    // prefetch tile k+1 into registers
            load16(fw, wptr + knext);
            // speculative L2 warm for tile k+2 (safe past end: TH=0 spec.)
            __builtin_prefetch(aptr + knext + TILE_K, 0, 0);
            __builtin_prefetch(wptr + knext + TILE_K, 0, 0);
            __syncthreads();             // LDS[buf] stores visible to all waves

            compute_tile(buf);

            // stage tile k+1 into the other buffer (no barrier needed:
            // nobody reads buf^1 until after the next iteration's barrier)
            cvt_store(&As[buf ^ 1][sdst], fa);
            cvt_store(&Ws[buf ^ 1][sdst], fw);
            buf ^= 1;
        }
        // peeled final tile: nothing left to stage
        __syncthreads();
        compute_tile(buf);
    }

    // ---- epilogue: C/D layout lane=(N%16, M half), VGPR r -> M = r + 8*hi
#pragma unroll
    for (int nf = 0; nf < 4; ++nf) {
        const int n = blockN + wn * 64 + nf * 16 + l16;
        float bv = bias0[n];
        if (HAS_B1) bv += bias1[n];
#pragma unroll
        for (int mf = 0; mf < 2; ++mf) {
#pragma unroll
            for (int r = 0; r < 8; ++r) {
                const size_t m = blockM + (size_t)(wm * 32 + mf * 16 + hi * 8 + r);
                C[m * (size_t)N + n] = acc[mf][nf][r] + bv;
            }
        }
    }
}

// fused = sigmoid(gacc)*g + (1-sigmoid)*cross; y = LN(fused + g)
__global__ __launch_bounds__(256)
void gate_ln_kernel(const float* __restrict__ g, const float* __restrict__ cross,
                    const float* __restrict__ gacc,
                    const float* __restrict__ ln_g, const float* __restrict__ ln_b,
                    float* __restrict__ out)
{
    const int F = 2048;
    const size_t base = (size_t)blockIdx.x * F;
    const int tid = threadIdx.x;

    float y[8];
    float s = 0.f, s2 = 0.f;
#pragma unroll
    for (int i = 0; i < 8; ++i) {
        const int c = tid + i * 256;
        const float gv = g[base + c];
        const float cv = cross[base + c];
        const float a  = gacc[base + c];
        const float gt = 1.0f / (1.0f + __expf(-a));
        const float yv = gt * gv + (1.0f - gt) * cv + gv;  // fused + residual
        y[i] = yv;
        s += yv;
        s2 += yv * yv;
    }
    // wave32 reduce, then cross-wave via LDS
#pragma unroll
    for (int off = 16; off > 0; off >>= 1) {
        s  += __shfl_xor(s,  off, 32);
        s2 += __shfl_xor(s2, off, 32);
    }
    __shared__ float ss[8], ss2[8];
    const int wv = tid >> 5, ln = tid & 31;
    if (ln == 0) { ss[wv] = s; ss2[wv] = s2; }
    __syncthreads();
    if (wv == 0) {
        float a = (ln < 8) ? ss[ln] : 0.f;
        float b = (ln < 8) ? ss2[ln] : 0.f;
#pragma unroll
        for (int off = 4; off > 0; off >>= 1) {
            a += __shfl_xor(a, off, 32);
            b += __shfl_xor(b, off, 32);
        }
        if (ln == 0) { ss[0] = a; ss2[0] = b; }
    }
    __syncthreads();
    const float mean = ss[0] * (1.0f / F);
    const float var  = ss2[0] * (1.0f / F) - mean * mean;
    const float inv  = rsqrtf(var + 1e-5f);
#pragma unroll
    for (int i = 0; i < 8; ++i) {
        const int c = tid + i * 256;
        out[base + c] = (y[i] - mean) * inv * ln_g[c] + ln_b[c];
    }
}

extern "C" void kernel_launch(void* const* d_in, const int* in_sizes, int n_in,
                              void* d_out, int out_size, void* d_ws, size_t ws_size,
                              hipStream_t stream) {
    (void)in_sizes; (void)n_in; (void)out_size; (void)ws_size;

    const float* graph    = (const float*)d_in[0];
    const float* temporal = (const float*)d_in[1];
    const float* gp_w  = (const float*)d_in[2];
    const float* gp_b  = (const float*)d_in[3];
    const float* tp_w  = (const float*)d_in[4];
    const float* tp_b  = (const float*)d_in[5];
    const float* in1_w = (const float*)d_in[6];
    const float* in1_b = (const float*)d_in[7];
    const float* out1_w = (const float*)d_in[8];
    const float* out1_b = (const float*)d_in[9];
    const float* in2_w = (const float*)d_in[10];
    const float* in2_b = (const float*)d_in[11];
    const float* out2_w = (const float*)d_in[12];
    const float* out2_b = (const float*)d_in[13];
    const float* gate_w = (const float*)d_in[14];
    const float* gate_b = (const float*)d_in[15];
    const float* ln_g  = (const float*)d_in[16];
    const float* ln_b  = (const float*)d_in[17];
    const float* lin_w = (const float*)d_in[18];
    const float* lin_b = (const float*)d_in[19];

    const int B = 16384, F = 2048, GD = 1024;
    const size_t E = (size_t)B * F;

    // workspace: 4 f32 buffers of B*F (512 MiB), reused across the pipeline
    float* g     = (float*)d_ws;   // live to the end
    float* t     = g + E;          // t -> v2 -> lnout
    float* v1    = t + E;          // v1 -> gacc
    float* cross = v1 + E;

    dim3 blk(256);
    dim3 grd(F / TILE_N, B / TILE_M);  // (16, 128)

    // projections
    wmma_gemm<1, false><<<grd, blk, 0, stream>>>(
        graph, gp_w, nullptr, nullptr, gp_b, nullptr, g, GD, GD);
    wmma_gemm<1, false><<<grd, blk, 0, stream>>>(
        temporal, tp_w, nullptr, nullptr, tp_b, nullptr, t, GD, GD);

    // v = kv @ Wv^T + bv  (Wv = rows 2F..3F of packed in_w; q/k are dead code)
    wmma_gemm<1, false><<<grd, blk, 0, stream>>>(
        t, in1_w + (size_t)2 * F * F, nullptr, nullptr, in1_b + 2 * F, nullptr, v1, F, F);
    float* v2 = t;  // t dead after v1 pass
    wmma_gemm<1, false><<<grd, blk, 0, stream>>>(
        g, in2_w + (size_t)2 * F * F, nullptr, nullptr, in2_b + 2 * F, nullptr, v2, F, F);

    // cross = v1 @ out1^T + b1 + v2 @ out2^T + b2  (fused dual-source GEMM)
    wmma_gemm<2, true><<<grd, blk, 0, stream>>>(
        v1, out1_w, v2, out2_w, out1_b, out2_b, cross, F, F);

    // gate logits = g @ gate_w[:, :F]^T + cross @ gate_w[:, F:]^T + gate_b
    float* gacc = v1;  // v1 dead after cross pass
    wmma_gemm<2, false><<<grd, blk, 0, stream>>>(
        g, gate_w, cross, gate_w + F, gate_b, nullptr, gacc, F, 2 * F);

    // sigmoid gate + residual + LayerNorm
    float* lnout = v2;  // v2 dead after cross pass
    gate_ln_kernel<<<dim3(B), blk, 0, stream>>>(g, cross, gacc, ln_g, ln_b, lnout);

    // final projection straight into d_out (f32)
    wmma_gemm<1, false><<<grd, blk, 0, stream>>>(
        lnout, lin_w, nullptr, nullptr, lin_b, nullptr, (float*)d_out, F, F);
}
